// LPN_90675349553472
// MI455X (gfx1250) — matrix operand order repaired and checked by reading, not compiled, not run on hardware
//
#include <hip/hip_runtime.h>
#include <hip/hip_bf16.h>

typedef __bf16 bf16_t;
typedef __attribute__((ext_vector_type(16))) __bf16 v16bf;
typedef __attribute__((ext_vector_type(8)))  float  v8f;

union FragBF16 {
    v16bf v;
    uint4 u[2];
};

#if defined(__has_builtin)
#if __has_builtin(__builtin_amdgcn_global_load_async_to_lds_b128)
#define LPN_HAVE_ASYNC_LDS 1
#endif
#if __has_builtin(__builtin_amdgcn_s_wait_asynccnt)
#define LPN_HAVE_WAIT_ASYNC_BUILTIN 1
#endif
#endif

#if defined(LPN_HAVE_ASYNC_LDS)
// Builtin signature (from hipcc diagnostic): (AS1 int4*, AS3 int4*, imm int, imm int)
typedef int lpn_v4i __attribute__((vector_size(16)));
typedef __attribute__((address_space(1))) lpn_v4i* lpn_gv4p;
typedef __attribute__((address_space(3))) lpn_v4i* lpn_lv4p;
#endif

__device__ __forceinline__ void lpn_wait_async0() {
#if defined(LPN_HAVE_WAIT_ASYNC_BUILTIN)
    __builtin_amdgcn_s_wait_asynccnt(0);
#elif defined(LPN_HAVE_ASYNC_LDS)
    asm volatile("s_wait_asynccnt 0" ::: "memory");
#endif
}

// ---------------------------------------------------------------------------
// Weight prep: HWIO f32 [3][3][CIN][192]  ->  [9][192][CIN] bf16 (B^T layout)
// ---------------------------------------------------------------------------
__global__ void lpn_prep_weights(const float* __restrict__ w,
                                 bf16_t* __restrict__ out, int CIN) {
    int i = blockIdx.x * blockDim.x + threadIdx.x;
    int total = 9 * 192 * CIN;
    if (i >= total) return;
    int tap = i / (192 * CIN);
    int rem = i - tap * 192 * CIN;
    int co  = rem / CIN;
    int ci  = rem - co * CIN;
    out[i] = (bf16_t)w[(tap * CIN + ci) * 192 + co];
}

// ---------------------------------------------------------------------------
// Implicit-GEMM 3x3 conv (SAME) + LayerNorm(bias) + ReLU, bf16 WMMA, f32 acc.
// Block: 256 threads = 8 waves. Output tile: 64 pixels (16 wide x 4 tall) x 192 ch.
// Wave (mg, ng): mg = 32-pixel group (2 image rows), ng = 48-channel group.
// Per k-step: 2 A-frags (4 ds_b128), 3 B-frags (6 global_b128), 6 WMMAs.
// LDS input tile padded to CIN+8 per pixel -> conflict-free ds_load_b128.
// ---------------------------------------------------------------------------
template <int CIN, bool IN_F32>
__global__ __launch_bounds__(256, 2)
void lpn_conv3x3_wmma(const void* __restrict__ in,     // [H][W][CIN] f32 or bf16
                      const bf16_t* __restrict__ wt,   // [9][192][CIN] bf16
                      const float* __restrict__ ln_b,  // [192]
                      bf16_t* __restrict__ out,        // [H][W][192]
                      int H, int W) {
    constexpr int COUT = 192;
    constexpr int CSTR = CIN + 8;                     // padded LDS pixel stride
    constexpr int IN_BYTES = 6 * 18 * CSTR * 2;       // halo tile bytes
    constexpr int OUT_BYTES = 64 * COUT * 4 + 512;    // f32 staging + LN stats
    constexpr int SMEM_BYTES = (IN_BYTES > OUT_BYTES) ? IN_BYTES : OUT_BYTES;
    constexpr int TOT = 6 * 18 * CIN;                 // logical halo elements

    __shared__ __align__(16) char smem[SMEM_BYTES];
    bf16_t* sin   = (bf16_t*)smem;
    float*  sout  = (float*)smem;                     // reused after K loop
    float*  sstat = (float*)(smem + 64 * COUT * 4);   // mean[64], rstd[64]

    const int tid = threadIdx.x;
    const int x0 = blockIdx.x * 16;
    const int y0 = blockIdx.y * 4;

    // ---- stage input halo tile (6 rows x 18 cols x CIN) into padded LDS ----
#if defined(LPN_HAVE_ASYNC_LDS)
    if constexpr (!IN_F32) {
        // bf16 input: per-lane async global->LDS b128 copies (8 elems/chunk)
        for (int e = tid * 8; e < TOT; e += 256 * 8) {
            const int pi = e / CIN;                   // halo pixel 0..107
            const int ci = e - pi * CIN;
            const int r = pi / 18, c = pi - r * 18;
            const int y = y0 + r - 1, x = x0 + c - 1;
            bf16_t* lp = sin + pi * CSTR + ci;
            if (y >= 0 && y < H && x >= 0 && x < W) {
                const bf16_t* gp = (const bf16_t*)in + ((size_t)y * W + x) * CIN + ci;
                __builtin_amdgcn_global_load_async_to_lds_b128(
                    (lpn_gv4p)gp, (lpn_lv4p)lp, 0, 0);
            } else {
                *(uint4*)lp = make_uint4(0u, 0u, 0u, 0u);
            }
        }
        lpn_wait_async0();
    } else
#endif
    {
        for (int e = tid * 4; e < TOT; e += 256 * 4) {
            const int pi = e / CIN;
            const int ci = e - pi * CIN;
            const int r = pi / 18, c = pi - r * 18;
            const int y = y0 + r - 1, x = x0 + c - 1;
            union { bf16_t h[4]; uint2 p; } t;
            if (y >= 0 && y < H && x >= 0 && x < W) {
                if (IN_F32) {
                    const float4 f = *(const float4*)((const float*)in + ((size_t)y * W + x) * CIN + ci);
                    t.h[0] = (bf16_t)f.x; t.h[1] = (bf16_t)f.y;
                    t.h[2] = (bf16_t)f.z; t.h[3] = (bf16_t)f.w;
                } else {
                    t.p = *(const uint2*)((const bf16_t*)in + ((size_t)y * W + x) * CIN + ci);
                }
            } else {
                t.h[0] = (bf16_t)0.f; t.h[1] = (bf16_t)0.f;
                t.h[2] = (bf16_t)0.f; t.h[3] = (bf16_t)0.f;
            }
            *(uint2*)(sin + pi * CSTR + ci) = t.p;
        }
    }

    // Hint the weight tensor into cache (global_prefetch_b8 path on gfx1250).
    __builtin_prefetch(wt + (size_t)tid * 128, 0, 0);

    __syncthreads();

    const int wave = tid >> 5;
    const int lane = tid & 31;
    const int mg   = wave & 1;    // pixel-row pair: rows {mg*2, mg*2+1}
    const int ng   = wave >> 1;   // 48-channel group: 0..3
    const int lm   = lane & 15;   // A: row M (=tx); B/C: column N
    const int lk   = lane >> 4;   // K-half selector

    v8f acc[6] = {};              // [mt][nt] -> acc[mt*3+nt]

    for (int tap = 0; tap < 9; ++tap) {
        const int dy = tap / 3, dx = tap - 3 * (tap / 3);
        const bf16_t* wtap = wt + (size_t)tap * COUT * CIN;
        const int arow0 = ((mg * 2 + 0 + dy) * 18 + lm + dx) * CSTR;
        const int arow1 = ((mg * 2 + 1 + dy) * 18 + lm + dx) * CSTR;
        for (int kc = 0; kc < CIN / 32; ++kc) {
            // A fragments 16x32 bf16: lane<16 K{0..7,16..23}, lane>=16 K{8..15,24..31}
            FragBF16 a0, a1;
            {
                const uint4* ap0 = (const uint4*)(sin + arow0 + kc * 32 + lk * 8);
                const uint4* ap1 = (const uint4*)(sin + arow1 + kc * 32 + lk * 8);
                a0.u[0] = ap0[0];
                a0.u[1] = ap0[2];  // +32B = +16 elements
                a1.u[0] = ap1[0];
                a1.u[1] = ap1[2];
            }
#pragma unroll
            for (int nt = 0; nt < 3; ++nt) {
                const int n = ng * 48 + nt * 16 + lm;
                // B fragment 32x16 bf16: lane holds column n, lane<16 K{0..7,16..23}, lane>=16 K{8..15,24..31}
                const uint4* bp = (const uint4*)(wtap + (size_t)n * CIN + kc * 32 + lk * 8);
                FragBF16 b;
                b.u[0] = bp[0];
                b.u[1] = bp[2];
                acc[0 * 3 + nt] = __builtin_amdgcn_wmma_f32_16x16x32_bf16(
                    false, a0.v, false, b.v, (short)0, acc[0 * 3 + nt], false, false);
                acc[1 * 3 + nt] = __builtin_amdgcn_wmma_f32_16x16x32_bf16(
                    false, a1.v, false, b.v, (short)0, acc[1 * 3 + nt], false, false);
            }
        }
    }

    __syncthreads();  // input tile no longer needed; reuse LDS for f32 staging

    // ---- scatter accumulators to LDS [pixel 0..63][channel 0..191] f32 ----
#pragma unroll
    for (int mt = 0; mt < 2; ++mt) {
#pragma unroll
        for (int nt = 0; nt < 3; ++nt) {
#pragma unroll
            for (int j = 0; j < 8; ++j) {
                const int p = (mg * 2 + mt) * 16 + j + lk * 8;  // C-layout: M = j + 8*(lane/16)
                const int n = ng * 48 + nt * 16 + lm;
                sout[p * COUT + n] = acc[mt * 3 + nt][j];
            }
        }
    }
    __syncthreads();

    // ---- per-pixel LayerNorm stats over 192 channels ----
    if (tid < 64) {
        float s = 0.f, ss = 0.f;
        for (int n = 0; n < COUT; ++n) {
            const float v = sout[tid * COUT + n];
            s += v; ss += v * v;
        }
        const float mu  = s * (1.0f / COUT);
        const float var = ss * (1.0f / COUT) - mu * mu;
        sstat[tid]      = mu;
        sstat[64 + tid] = rsqrtf(var + 1e-6f);
    }
    __syncthreads();

    // ---- normalize + bias + ReLU, convert to bf16, store ----
    for (int e = tid; e < 64 * COUT; e += 256) {
        const int p = e / COUT;
        const int n = e - p * COUT;
        float v = (sout[e] - sstat[p]) * sstat[64 + p] + ln_b[n];
        v = v > 0.f ? v : 0.f;
        const int ty = p >> 4, tx = p & 15;
        out[((size_t)(y0 + ty) * W + (x0 + tx)) * COUT + n] = (bf16_t)v;
    }
}

// ---------------------------------------------------------------------------
// Head: 1x1 convs (192->2 cls, 192->2 reg) + softmax + grid/validity/scale
// ---------------------------------------------------------------------------
__global__ void lpn_head(const bf16_t* __restrict__ act,
                         const float* __restrict__ w_cls, const float* __restrict__ b_cls,
                         const float* __restrict__ w_reg, const float* __restrict__ b_reg,
                         float* __restrict__ scores, float* __restrict__ locs,
                         int* __restrict__ cls,
                         int H, int W, float scale, int off) {
    const int i = blockIdx.x * blockDim.x + threadIdx.x;
    if (i >= H * W) return;
    const bf16_t* xp = act + (size_t)i * 192;
    float l0 = b_cls[0], l1 = b_cls[1];
    float r0 = b_reg[0], r1 = b_reg[1];
    for (int c = 0; c < 192; ++c) {
        const float xv = (float)xp[c];
        l0 += xv * w_cls[c * 2 + 0];
        l1 += xv * w_cls[c * 2 + 1];
        r0 += xv * w_reg[c * 2 + 0];
        r1 += xv * w_reg[c * 2 + 1];
    }
    const float m  = fmaxf(l0, l1);
    const float e0 = expf(l0 - m), e1 = expf(l1 - m);
    const float p0 = e0 / (e0 + e1);  // foreground prob (N_CLS == 1)
    const int y = i / W, x = i - (i / W) * W;
    const float ly = (float)y + 0.5f + r0;
    const float lx = (float)x + 0.5f + r1;
    const bool valid = (ly > 0.f) && (lx > 0.f) && (ly < (float)H) && (lx < (float)W);
    scores[off + i]           = valid ? p0 : -1.0f;
    locs[(size_t)(off + i) * 2 + 0] = ly * scale;
    locs[(size_t)(off + i) * 2 + 1] = lx * scale;
    cls[off + i] = 0;
}

// ---------------------------------------------------------------------------
// Top-k selection: histogram -> threshold -> compact -> bitonic sort (4096)
// ---------------------------------------------------------------------------
__global__ void lpn_init_small(int* hist, int* count) {
    const int i = blockIdx.x * blockDim.x + threadIdx.x;
    if (i < 1024) hist[i] = 0;
    if (i == 0) count[0] = 0;
}

__global__ void lpn_hist(const float* __restrict__ scores, int N, int* hist) {
    const int i = blockIdx.x * blockDim.x + threadIdx.x;
    if (i >= N) return;
    const float s = scores[i];
    if (s > 0.2f) {
        int b = (int)((s - 0.2f) * (1024.0f / 0.8f));
        b = b < 0 ? 0 : (b > 1023 ? 1023 : b);
        atomicAdd(&hist[b], 1);
    }
}

__global__ void lpn_thresh(const int* __restrict__ hist, float* thr) {
    int cum = 0;
    float t = 0.2f;
    for (int b = 1023; b >= 0; --b) {
        cum += hist[b];
        if (cum >= 2048) { t = 0.2f + (float)b * (0.8f / 1024.0f); break; }
    }
    thr[0] = t;
}

__global__ void lpn_compact(const float* __restrict__ scores,
                            const float* __restrict__ locs,
                            const int* __restrict__ cls, int N,
                            const float* __restrict__ thr, int* count,
                            float* cs, float* cl0, float* cl1, int* ccl) {
    const int i = blockIdx.x * blockDim.x + threadIdx.x;
    if (i >= N) return;
    const float s = scores[i];
    if (s > thr[0] && s > 0.2f) {
        const int slot = atomicAdd(count, 1);
        if (slot < 4096) {
            cs[slot]  = s;
            cl0[slot] = locs[(size_t)i * 2 + 0];
            cl1[slot] = locs[(size_t)i * 2 + 1];
            ccl[slot] = cls[i];
        }
    }
}

__global__ __launch_bounds__(1024)
void lpn_sort(const int* __restrict__ count,
              const float* __restrict__ cs, const float* __restrict__ cl0,
              const float* __restrict__ cl1, const int* __restrict__ ccl,
              float* ss, float* sl0, float* sl1, int* scl) {
    __shared__ float sk[4096];
    __shared__ int   sv[4096];
    const int tid = threadIdx.x;
    const int m = count[0] < 4096 ? count[0] : 4096;
    for (int i = tid; i < 4096; i += 1024) {
        sk[i] = (i < m) ? cs[i] : -1e30f;
        sv[i] = i;
    }
    __syncthreads();
    // bitonic sort, descending
    for (int k = 2; k <= 4096; k <<= 1) {
        for (int j = k >> 1; j > 0; j >>= 1) {
            for (int i = tid; i < 4096; i += 1024) {
                const int ixj = i ^ j;
                if (ixj > i) {
                    const bool upper = ((i & k) == 0);
                    const float a = sk[i], b = sk[ixj];
                    const bool swap = upper ? (a < b) : (a > b);
                    if (swap) {
                        sk[i] = b; sk[ixj] = a;
                        const int tv = sv[i]; sv[i] = sv[ixj]; sv[ixj] = tv;
                    }
                }
            }
            __syncthreads();
        }
    }
    for (int i = tid; i < 2048; i += 1024) {
        const float s = sk[i];
        if (i < m && s > -1e29f) {
            const int v = sv[i];
            ss[i] = s; sl0[i] = cl0[v]; sl1[i] = cl1[v]; scl[i] = ccl[v];
        } else {
            ss[i] = -2.0f; sl0[i] = 0.f; sl1[i] = 0.f; scl[i] = 0;
        }
    }
}

// ---------------------------------------------------------------------------
// Serial sorted-NMS, single wave32. dist_sq threshold = NMS_THRESHOLD^2 = 64.
// d_out = [scores 1280][locs 1280*2][cls 1280], all f32, default -1.
// ---------------------------------------------------------------------------
__global__ void lpn_nms(const float* __restrict__ ss, const float* __restrict__ sl0,
                        const float* __restrict__ sl1, const int* __restrict__ scl,
                        float* __restrict__ out) {
    __shared__ float k0[1280], k1[1280];
    const int lane = threadIdx.x;
    for (int i = lane; i < 1280 * 4; i += 32) out[i] = -1.0f;
    __syncthreads();
    int cnt = 0;
    for (int i = 0; i < 2048; ++i) {
        const float s = ss[i];
        if (s <= 0.2f) continue;
        if (cnt >= 1280) break;
        const float a = sl0[i], b = sl1[i];
        bool sup = false;
        for (int j = lane; j < cnt; j += 32) {
            const float d0 = k0[j] - a, d1 = k1[j] - b;
            if (d0 * d0 + d1 * d1 < 64.0f) sup = true;
        }
        const bool any_sup = (__ballot((int)sup) != 0ull);
        if (!any_sup) {
            if (lane == 0) {
                k0[cnt] = a; k1[cnt] = b;
                out[cnt]                  = s;
                out[1280 + cnt * 2 + 0]   = a;
                out[1280 + cnt * 2 + 1]   = b;
                out[1280 * 3 + cnt]       = (float)scl[i];
            }
            cnt++;
        }
        __syncthreads();
    }
}

// ---------------------------------------------------------------------------
// Launch: d_in order = feat0, feat1, feat2, then per level (keys sorted):
//   b_cls, b_reg, ln_b0..3, w0..w3, w_cls, w_reg
// ---------------------------------------------------------------------------
extern "C" void kernel_launch(void* const* d_in, const int* in_sizes, int n_in,
                              void* d_out, int out_size, void* d_ws, size_t ws_size,
                              hipStream_t stream) {
    (void)in_sizes; (void)n_in; (void)out_size; (void)ws_size;
    const float* feat[3] = {(const float*)d_in[0], (const float*)d_in[1], (const float*)d_in[2]};
    const int   HWs[3]    = {256, 128, 64};
    const float scales[3] = {4.f, 8.f, 16.f};

    char* ws = (char*)d_ws;
    size_t off = 0;
    auto alloc = [&](size_t bytes) -> char* {
        char* p = ws + off;
        off += (bytes + 255) & ~(size_t)255;
        return p;
    };

    bf16_t* actA = (bf16_t*)alloc((size_t)256 * 256 * 192 * 2);
    bf16_t* actB = (bf16_t*)alloc((size_t)256 * 256 * 192 * 2);
    bf16_t* wtb[3][4];
    for (int l = 0; l < 3; ++l) {
        wtb[l][0] = (bf16_t*)alloc((size_t)9 * 192 * 256 * 2);
        for (int i = 1; i < 4; ++i) wtb[l][i] = (bf16_t*)alloc((size_t)9 * 192 * 192 * 2);
    }
    const int NTOT = 256 * 256 + 128 * 128 + 64 * 64;  // 86016
    float* scoresBuf = (float*)alloc((size_t)NTOT * 4);
    float* locBuf    = (float*)alloc((size_t)NTOT * 2 * 4);
    int*   clsBuf    = (int*)alloc((size_t)NTOT * 4);
    int*   hist      = (int*)alloc(1024 * 4);
    int*   count     = (int*)alloc(64);
    float* thr       = (float*)alloc(64);
    float* cs  = (float*)alloc(4096 * 4);
    float* cl0 = (float*)alloc(4096 * 4);
    float* cl1 = (float*)alloc(4096 * 4);
    int*   ccl = (int*)alloc(4096 * 4);
    float* ss  = (float*)alloc(2048 * 4);
    float* sl0 = (float*)alloc(2048 * 4);
    float* sl1 = (float*)alloc(2048 * 4);
    int*   scl = (int*)alloc(2048 * 4);

    int pixoff = 0;
    for (int lvl = 0; lvl < 3; ++lvl) {
        const int base = 3 + 12 * lvl;
        const float* b_cls = (const float*)d_in[base + 0];
        const float* b_reg = (const float*)d_in[base + 1];
        const float* lnb[4] = {(const float*)d_in[base + 2], (const float*)d_in[base + 3],
                               (const float*)d_in[base + 4], (const float*)d_in[base + 5]};
        const float* wconv[4] = {(const float*)d_in[base + 6], (const float*)d_in[base + 7],
                                 (const float*)d_in[base + 8], (const float*)d_in[base + 9]};
        const float* w_cls = (const float*)d_in[base + 10];
        const float* w_reg = (const float*)d_in[base + 11];

        {
            const int tot0 = 9 * 192 * 256;
            lpn_prep_weights<<<(tot0 + 255) / 256, 256, 0, stream>>>(wconv[0], wtb[lvl][0], 256);
            const int tot1 = 9 * 192 * 192;
            for (int i = 1; i < 4; ++i)
                lpn_prep_weights<<<(tot1 + 255) / 256, 256, 0, stream>>>(wconv[i], wtb[lvl][i], 192);
        }

        const int H = HWs[lvl], W = HWs[lvl];
        dim3 grid(W / 16, H / 4);
        lpn_conv3x3_wmma<256, true ><<<grid, 256, 0, stream>>>(feat[lvl], wtb[lvl][0], lnb[0], actA, H, W);
        lpn_conv3x3_wmma<192, false><<<grid, 256, 0, stream>>>(actA,      wtb[lvl][1], lnb[1], actB, H, W);
        lpn_conv3x3_wmma<192, false><<<grid, 256, 0, stream>>>(actB,      wtb[lvl][2], lnb[2], actA, H, W);
        lpn_conv3x3_wmma<192, false><<<grid, 256, 0, stream>>>(actA,      wtb[lvl][3], lnb[3], actB, H, W);

        lpn_head<<<(H * W + 255) / 256, 256, 0, stream>>>(actB, w_cls, b_cls, w_reg, b_reg,
                                                          scoresBuf, locBuf, clsBuf,
                                                          H, W, scales[lvl], pixoff);
        pixoff += H * W;
    }

    lpn_init_small<<<4, 256, 0, stream>>>(hist, count);
    lpn_hist<<<(NTOT + 255) / 256, 256, 0, stream>>>(scoresBuf, NTOT, hist);
    lpn_thresh<<<1, 1, 0, stream>>>(hist, thr);
    lpn_compact<<<(NTOT + 255) / 256, 256, 0, stream>>>(scoresBuf, locBuf, clsBuf, NTOT,
                                                        thr, count, cs, cl0, cl1, ccl);
    lpn_sort<<<1, 1024, 0, stream>>>(count, cs, cl0, cl1, ccl, ss, sl0, sl1, scl);
    lpn_nms<<<1, 32, 0, stream>>>(ss, sl0, sl1, scl, (float*)d_out);
}
